// SkeletonMamba_20504173871377
// MI455X (gfx1250) — matrix operand order repaired and checked by reading, not compiled, not run on hardware
//
#include <hip/hip_runtime.h>
#include <hip/hip_bf16.h>

// SkeletonMamba for gfx1250 (MI455X). All GEMMs on the WMMA pipe
// (v_wmma_f32_16x16x32_f16), scan/activations in f32 VALU.
// Problem fits in L2 (~22MB) -> compute/latency bound, not HBM bound.
// GEMM buffers are row-padded to 1360 (=85*16) so WMMA tiles need no
// EXEC-divergent guards; per-lane fragments load as contiguous b128s.
// K is a template parameter -> fully unrolled straight-line GEMM body so
// the scheduler can hoist loads and use partial s_wait_loadcnt waits.

#define Bn 2
#define Tn 16
#define Jn 42
#define Ln 673           // T*J + 1
#define BL 1346          // B * L
#define BLP 1360         // BL padded to 16-row tiles (85 tiles)
#define DM 192           // d_model
#define DI 384           // d_inner
#define DS 16            // d_state
#define DEPTHn 16

typedef __attribute__((ext_vector_type(16))) _Float16 v16h;
typedef __attribute__((ext_vector_type(8)))  _Float16 v8h;
typedef __attribute__((ext_vector_type(8)))  float    v8f;

// ---------------------------------------------------------------------------
// f32 -> f16 weight conversion with zero-padding: dst[(l*Npad+n)*Kpad+k]
// ---------------------------------------------------------------------------
__global__ void sm_convert_pad(const float* __restrict__ src, _Float16* __restrict__ dst,
                               int layers, int Nsrc, int Ksrc, int Npad, int Kpad) {
    int e = blockIdx.x * blockDim.x + threadIdx.x;
    int total = layers * Npad * Kpad;
    if (e >= total) return;
    int k = e % Kpad;
    int n = (e / Kpad) % Npad;
    int l = e / (Kpad * Npad);
    float v = (n < Nsrc && k < Ksrc) ? src[((size_t)l * Nsrc + n) * Ksrc + k] : 0.0f;
    dst[e] = (_Float16)v;
}

// ---------------------------------------------------------------------------
// Embedding: joint proj (K=3) + positional encodings + cls token.
// Also zeroes the residual stream.
// ---------------------------------------------------------------------------
__global__ void sm_embed(const float* __restrict__ x, const float* __restrict__ jW,
                         const float* __restrict__ jb, const float* __restrict__ cls,
                         const float* __restrict__ spos, const float* __restrict__ tpos,
                         float* __restrict__ hidden, float* __restrict__ residual) {
    int e = blockIdx.x * blockDim.x + threadIdx.x;
    if (e >= Bn * Ln * DM) return;
    int d = e % DM;
    int l = (e / DM) % Ln;
    int b = e / (Ln * DM);
    float val;
    if (l == 0) {
        val = cls[d] + spos[d];
    } else {
        int lp = l - 1;
        int t = lp / Jn;
        int j = lp % Jn;
        const float* xp = &x[(((size_t)b * Tn + t) * Jn + j) * 3];
        val = jb[d] + xp[0] * jW[d * 3 + 0] + xp[1] * jW[d * 3 + 1] + xp[2] * jW[d * 3 + 2]
            + spos[(j + 1) * DM + d] + tpos[t * DM + d];
    }
    hidden[(size_t)(b * Ln + l) * DM + d] = val;
    residual[(size_t)(b * Ln + l) * DM + d] = 0.0f;
}

// ---------------------------------------------------------------------------
// residual += hidden; hs16 = f16(rmsnorm(residual) * w)   (one block per row)
// ---------------------------------------------------------------------------
__global__ void sm_prenorm(const float* __restrict__ hidden, float* __restrict__ residual,
                           _Float16* __restrict__ hs16, const float* __restrict__ nw) {
    __shared__ float s[DM];
    __shared__ float inv;
    int r = blockIdx.x;
    int d = threadIdx.x;
    float v = hidden[(size_t)r * DM + d] + residual[(size_t)r * DM + d];
    residual[(size_t)r * DM + d] = v;
    s[d] = v * v;
    __syncthreads();
    if (d < 64) s[d] += s[d + 128];
    __syncthreads();
    if (d < 64) s[d] += s[d + 64];
    __syncthreads();
    if (d < 32) s[d] += s[d + 32];
    __syncthreads();
    if (d == 0) {
        float t = 0.0f;
        for (int i = 0; i < 32; ++i) t += s[i];
        inv = rsqrtf(t / (float)DM + 1e-5f);
    }
    __syncthreads();
    hs16[(size_t)r * DM + d] = (_Float16)(v * inv * nw[d]);
}

// ---------------------------------------------------------------------------
// WMMA GEMM: C[BLP,N] (f32) = A[BLP,K] (f16 row-major) x W[N,K]^T (f16 row-major)
// 4 waves/block; each wave computes a 16x(16*NTPW) strip with NTPW independent
// accumulators, reusing one A fragment per k-step. K (compile-time) % 32 == 0.
// Fully unrolled: one straight-line block of loads+WMMAs for max ILP.
//   A frag: lane m=lane&15, halves [kb+hi*8, +7] and [kb+16+hi*8, +7]
//   B frag: lane n=lane&15, halves [kb+hi*8, +7] and [kb+16+hi*8, +7]
//   C frag: VGPR r -> row r+hi*8, col lane&15
// ---------------------------------------------------------------------------
template <int NTPW, int K>
__global__ void sm_gemm_wmma(const _Float16* __restrict__ A, const _Float16* __restrict__ W,
                             float* __restrict__ C, int N) {
    int lane   = threadIdx.x & 31;
    int wave   = threadIdx.x >> 5;
    int NT     = N >> 4;
    int nTile0 = (blockIdx.x * 4 + wave) * NTPW;
    int mTile  = blockIdx.y;
    if (nTile0 >= NT) return;                 // wave-uniform; EXEC stays full

    int hi   = (lane >= 16) ? 1 : 0;
    int mrow = mTile * 16 + (lane & 15);
    int nclo = lane & 15;

    v8f zero = {};
    v8f acc[NTPW];
    #pragma unroll
    for (int j = 0; j < NTPW; ++j) acc[j] = zero;

    const _Float16* Arow = A + (size_t)mrow * K + hi * 8;
    #pragma unroll
    for (int kb = 0; kb < K; kb += 32) {
        v8h alo = *(const v8h*)(Arow + kb);        // 16B -> global_load_b128
        v8h ahi = *(const v8h*)(Arow + kb + 16);   // 16B -> global_load_b128
        v16h av;
        #pragma unroll
        for (int i = 0; i < 8; ++i) { av[i] = alo[i]; av[i + 8] = ahi[i]; }
        #pragma unroll
        for (int j = 0; j < NTPW; ++j) {
            const _Float16* Wrow = W + (size_t)((nTile0 + j) * 16 + nclo) * K + kb + hi * 8;
            v8h blo = *(const v8h*)(Wrow);
            v8h bhi = *(const v8h*)(Wrow + 16);
            v16h bv;
            #pragma unroll
            for (int i = 0; i < 8; ++i) { bv[i] = blo[i]; bv[i + 8] = bhi[i]; }
            acc[j] = __builtin_amdgcn_wmma_f32_16x16x32_f16(
                /*neg_a=*/false, av, /*neg_b=*/false, bv,
                /*c_mod=*/(short)0, acc[j], /*reuse_a=*/false, /*reuse_b=*/false);
        }
    }
    #pragma unroll
    for (int j = 0; j < NTPW; ++j) {
        float* Crow = C + (size_t)(mTile * 16 + hi * 8) * N + (nTile0 + j) * 16 + nclo;
        #pragma unroll
        for (int r = 0; r < 8; ++r) Crow[(size_t)r * N] = acc[j][r];
    }
}

// ---------------------------------------------------------------------------
// Depthwise causal conv (width 4) over sequence + bias + SiLU.
// Reads x half of xz; writes f32 (for scan) and f16 (for x_proj GEMM).
// ---------------------------------------------------------------------------
__global__ void sm_conv_silu(const float* __restrict__ xz, const float* __restrict__ cW,
                             const float* __restrict__ cb, float* __restrict__ xf,
                             _Float16* __restrict__ x16) {
    int e = blockIdx.x * blockDim.x + threadIdx.x;
    if (e >= BL * DI) return;
    int d = e % DI;
    int l = (e / DI) % Ln;
    int b = e / (Ln * DI);
    float acc = cb[d];
    #pragma unroll
    for (int k = 0; k < 4; ++k) {
        int lp = l - 3 + k;
        if (lp >= 0) acc += xz[((size_t)(b * Ln + lp)) * (2 * DI) + d] * cW[d * 4 + k];
    }
    float sv = acc / (1.0f + expf(-acc));    // SiLU
    xf[e]  = sv;
    x16[e] = (_Float16)sv;
}

// ---------------------------------------------------------------------------
// Split dbl[BLP,48] -> dt_low (f16, padded to 32), Bc[BL,16], Cc[BL,16]
// ---------------------------------------------------------------------------
__global__ void sm_split_dbl(const float* __restrict__ dbl, _Float16* __restrict__ dtlow16,
                             float* __restrict__ Bc, float* __restrict__ Cc) {
    int e = blockIdx.x * blockDim.x + threadIdx.x;
    if (e >= BL * 64) return;
    int r = e / 64;
    int i = e % 64;
    if (i < 32) {
        float v = (i < 12) ? dbl[(size_t)r * 48 + i] : 0.0f;
        dtlow16[(size_t)r * 32 + i] = (_Float16)v;
    } else if (i < 48) {
        Bc[(size_t)r * DS + (i - 32)] = dbl[(size_t)r * 48 + 12 + (i - 32)];
    } else {
        Cc[(size_t)r * DS + (i - 48)] = dbl[(size_t)r * 48 + 28 + (i - 48)];
    }
}

// ---------------------------------------------------------------------------
// dt = softplus(dt_pre + dt_bias)   (in place, valid rows only)
// ---------------------------------------------------------------------------
__global__ void sm_softplus(float* __restrict__ dt, const float* __restrict__ dtb) {
    int e = blockIdx.x * blockDim.x + threadIdx.x;
    if (e >= BL * DI) return;
    int d = e % DI;
    float v = dt[e] + dtb[d];
    dt[e] = (v > 20.0f) ? v : log1pf(expf(v));
}

// ---------------------------------------------------------------------------
// Selective scan: one thread per (b, d_inner); 16 states in registers.
// Fuses dA/dBx, D-skip, SiLU(z) gate; writes f16 for out_proj GEMM.
// ---------------------------------------------------------------------------
__global__ void sm_scan(const float* __restrict__ dt, const float* __restrict__ xf,
                        const float* __restrict__ xz, const float* __restrict__ Bc,
                        const float* __restrict__ Cc, const float* __restrict__ A_log,
                        const float* __restrict__ Dp, _Float16* __restrict__ y16) {
    int tid = blockIdx.x * blockDim.x + threadIdx.x;
    if (tid >= Bn * DI) return;
    int b = tid / DI;
    int d = tid % DI;
    float a[DS], h[DS];
    #pragma unroll
    for (int n = 0; n < DS; ++n) {
        a[n] = -expf(A_log[d * DS + n]);
        h[n] = 0.0f;
    }
    float Dv = Dp[d];
    for (int t = 0; t < Ln; ++t) {
        int r = b * Ln + t;
        float dtv = dt[(size_t)r * DI + d];
        float xv  = xf[(size_t)r * DI + d];
        float zv  = xz[(size_t)r * (2 * DI) + DI + d];
        const float* Bp = &Bc[(size_t)r * DS];
        const float* Cp = &Cc[(size_t)r * DS];
        float yv = 0.0f;
        #pragma unroll
        for (int n = 0; n < DS; ++n) {
            h[n] = expf(dtv * a[n]) * h[n] + dtv * Bp[n] * xv;
            yv  += h[n] * Cp[n];
        }
        yv += Dv * xv;
        float g = zv / (1.0f + expf(-zv));   // SiLU(z)
        y16[(size_t)r * DI + d] = (_Float16)(yv * g);
    }
}

// ---------------------------------------------------------------------------
// Final rmsnorm (token 0 only) + classifier head. One block per batch.
// ---------------------------------------------------------------------------
__global__ void sm_final_head(const float* __restrict__ hidden, const float* __restrict__ residual,
                              const float* __restrict__ nfw, const float* __restrict__ hW,
                              const float* __restrict__ hb, float* __restrict__ out) {
    __shared__ float nv[DM];
    __shared__ float red[256];
    int b   = blockIdx.x;
    int tid = threadIdx.x;
    size_t r0 = (size_t)b * Ln * DM;
    float v = 0.0f;
    if (tid < DM) v = hidden[r0 + tid] + residual[r0 + tid];
    red[tid] = v * v;
    __syncthreads();
    for (int s = 128; s > 0; s >>= 1) {
        if (tid < s) red[tid] += red[tid + s];
        __syncthreads();
    }
    float inv = rsqrtf(red[0] / (float)DM + 1e-5f);
    if (tid < DM) nv[tid] = v * inv * nfw[tid];
    __syncthreads();
    for (int c = tid; c < 1000; c += 256) {
        float acc = hb[c];
        for (int d = 0; d < DM; ++d) acc += nv[d] * hW[(size_t)c * DM + d];
        out[(size_t)b * 1000 + c] = acc;
    }
}

// ---------------------------------------------------------------------------
extern "C" void kernel_launch(void* const* d_in, const int* in_sizes, int n_in,
                              void* d_out, int out_size, void* d_ws, size_t ws_size,
                              hipStream_t stream) {
    const float* x_in      = (const float*)d_in[0];
    const float* joint_W   = (const float*)d_in[1];
    const float* joint_b   = (const float*)d_in[2];
    const float* cls_token = (const float*)d_in[3];
    const float* spos      = (const float*)d_in[4];
    const float* tpos      = (const float*)d_in[5];
    const float* norm_w    = (const float*)d_in[6];
    const float* in_proj_W = (const float*)d_in[7];
    const float* conv_W    = (const float*)d_in[8];
    const float* conv_b    = (const float*)d_in[9];
    const float* x_proj_W  = (const float*)d_in[10];
    const float* dt_proj_W = (const float*)d_in[11];
    const float* dt_proj_b = (const float*)d_in[12];
    const float* A_log     = (const float*)d_in[13];
    const float* D_param   = (const float*)d_in[14];
    const float* out_projW = (const float*)d_in[15];
    const float* norm_f_w  = (const float*)d_in[16];
    const float* head_W    = (const float*)d_in[17];
    const float* head_b    = (const float*)d_in[18];
    float* out = (float*)d_out;

    // ---- workspace carve-up (~22 MB total); GEMM-facing buffers padded to BLP
    char* base = (char*)d_ws;
    size_t off = 0;
    auto alloc = [&](size_t bytes) -> char* {
        char* p = base + off;
        off += (bytes + 255) & ~(size_t)255;
        return p;
    };
    float*    hidden   = (float*)   alloc((size_t)BLP * DM * 4);       // C of out_proj
    float*    residual = (float*)   alloc((size_t)BL  * DM * 4);
    _Float16* hs16     = (_Float16*)alloc((size_t)BLP * DM * 2);       // A of in_proj
    float*    xz       = (float*)   alloc((size_t)BLP * 2 * DI * 4);   // C of in_proj
    float*    xf       = (float*)   alloc((size_t)BL  * DI * 4);
    _Float16* x16      = (_Float16*)alloc((size_t)BLP * DI * 2);       // A of x_proj
    float*    dbl      = (float*)   alloc((size_t)BLP * 48 * 4);       // C of x_proj
    _Float16* dtlow16  = (_Float16*)alloc((size_t)BLP * 32 * 2);       // A of dt_proj
    float*    Bc       = (float*)   alloc((size_t)BL  * DS * 4);
    float*    Cc       = (float*)   alloc((size_t)BL  * DS * 4);
    float*    dt       = (float*)   alloc((size_t)BLP * DI * 4);       // C of dt_proj
    _Float16* y16      = (_Float16*)alloc((size_t)BLP * DI * 2);       // A of out_proj
    _Float16* inW16    = (_Float16*)alloc((size_t)DEPTHn * 768 * DM * 2);
    _Float16* xpW16    = (_Float16*)alloc((size_t)DEPTHn * 48 * DI * 2);
    _Float16* dtW16    = (_Float16*)alloc((size_t)DEPTHn * DI * 32 * 2);
    _Float16* outW16   = (_Float16*)alloc((size_t)DEPTHn * DM * DI * 2);

    auto cdiv = [](int a, int b) { return (a + b - 1) / b; };

    // ---- weight conversion (f32 -> f16, zero-padded) ----
    {
        int t;
        t = DEPTHn * 768 * DM;
        sm_convert_pad<<<cdiv(t, 256), 256, 0, stream>>>(in_proj_W, inW16, DEPTHn, 768, DM, 768, DM);
        t = DEPTHn * 48 * DI;
        sm_convert_pad<<<cdiv(t, 256), 256, 0, stream>>>(x_proj_W, xpW16, DEPTHn, 44, DI, 48, DI);
        t = DEPTHn * DI * 32;
        sm_convert_pad<<<cdiv(t, 256), 256, 0, stream>>>(dt_proj_W, dtW16, DEPTHn, DI, 12, DI, 32);
        t = DEPTHn * DM * DI;
        sm_convert_pad<<<cdiv(t, 256), 256, 0, stream>>>(out_projW, outW16, DEPTHn, DM, DI, DM, DI);
    }

    // ---- embedding ----
    sm_embed<<<cdiv(Bn * Ln * DM, 256), 256, 0, stream>>>(
        x_in, joint_W, joint_b, cls_token, spos, tpos, hidden, residual);

    const int MT = BLP / 16;   // 85 row tiles

    // ---- layer loop ----
    for (int l = 0; l < DEPTHn; ++l) {
        const _Float16* inW  = inW16  + (size_t)l * 768 * DM;
        const _Float16* xpW  = xpW16  + (size_t)l * 48 * DI;
        const _Float16* dtW  = dtW16  + (size_t)l * DI * 32;
        const _Float16* outW = outW16 + (size_t)l * DM * DI;

        sm_prenorm<<<BL, DM, 0, stream>>>(hidden, residual, hs16, norm_w + (size_t)l * DM);

        // xz = hs @ in_proj^T : [BLP,192] x [768,192]^T -> [BLP,768]  (NT=48)
        sm_gemm_wmma<4, DM><<<dim3(cdiv(48, 16), MT), 128, 0, stream>>>(hs16, inW, xz, 768);

        sm_conv_silu<<<cdiv(BL * DI, 256), 256, 0, stream>>>(
            xz, conv_W + (size_t)l * DI * 4, conv_b + (size_t)l * DI, xf, x16);

        // dbl = x @ x_proj^T : [BLP,384] x [48,384]^T -> [BLP,48]  (NT=3)
        sm_gemm_wmma<1, DI><<<dim3(1, MT), 128, 0, stream>>>(x16, xpW, dbl, 48);

        sm_split_dbl<<<cdiv(BL * 64, 256), 256, 0, stream>>>(dbl, dtlow16, Bc, Cc);

        // dt_pre = dt_low @ dt_proj^T : [BLP,32] x [384,32]^T -> [BLP,384]  (NT=24)
        sm_gemm_wmma<4, 32><<<dim3(cdiv(24, 16), MT), 128, 0, stream>>>(dtlow16, dtW, dt, DI);

        sm_softplus<<<cdiv(BL * DI, 256), 256, 0, stream>>>(dt, dt_proj_b + (size_t)l * DI);

        sm_scan<<<cdiv(Bn * DI, 256), 256, 0, stream>>>(
            dt, xf, xz, Bc, Cc, A_log + (size_t)l * DI * DS, D_param + (size_t)l * DI, y16);

        // hidden = y @ out_proj^T : [BLP,384] x [192,384]^T -> [BLP,192]  (NT=12)
        sm_gemm_wmma<4, DI><<<dim3(1, MT), 128, 0, stream>>>(y16, outW, hidden, DM);
    }

    // ---- final norm (token 0) + head ----
    sm_final_head<<<Bn, 256, 0, stream>>>(hidden, residual, norm_f_w, head_W, head_b, out);
}
